// Network_87385404605020
// MI455X (gfx1250) — compile-verified, matrix operand order checked by reading, not run on hardware
//
#include <hip/hip_runtime.h>
#include <hip/hip_bf16.h>

typedef __attribute__((ext_vector_type(16))) _Float16 v16h;
typedef __attribute__((ext_vector_type(8)))  _Float16 v8h;
typedef __attribute__((ext_vector_type(8)))  float    v8f;
typedef __attribute__((ext_vector_type(4)))  float    v4f;

#define THETA    5120.0f            /* 80 * 64 */
#define I_DECAY  0.75f              /* (4096-1024)/4096 */
#define V_DECAY  0.96875f           /* (4096-128)/4096  */
#define POOL_WT  5632.0f            /* 1.1 * THETA      */
#define TSTEPS   50

// CDNA5 16-bit 16x32 A/B fragment: K index held by (lane, element i).
// VGPR0-3: lanes 0-15 K=0..7, lanes 16-31 K=8..15; VGPR4-7: +16.
__device__ __forceinline__ int kmap(int lane, int i) {
    return ((i & 8) << 1) | ((lane & 16) >> 1) | (i & 7);
}

// ---------------------------------------------------------------------------
// Stage A: 4x4 sum-pool on raw DVS events (layout [N,2,400,304,T], T inner)
// fused with CUBA neuron scan + delay; spikes out as f16, layout [N,T,C,H,W].
// ---------------------------------------------------------------------------
__global__ void pool4_loihi_kernel(const float* __restrict__ X,
                                   _Float16* __restrict__ S) {
    const int H = 100, W = 76, C = 2;
    int nid = blockIdx.x * blockDim.x + threadIdx.x;
    if (nid >= 2 * C * H * W) return;
    int w = nid % W, h = (nid / W) % H, c = (nid / (W * H)) % C, n = nid / (W * H * C);
    const float* xb = X + (((size_t)(n * C + c) * 400 + h * 4) * 304 + (size_t)w * 4) * TSTEPS;
    const int CHW = C * H * W;
    _Float16* sb = S + (size_t)n * TSTEPS * CHW + (size_t)c * H * W + h * W + w;
    sb[0] = (_Float16)0.f;
    float u = 0.f, v = 0.f;
    for (int t = 0; t < TSTEPS; ++t) {
        float sum = 0.f;
#pragma unroll
        for (int dy = 0; dy < 4; ++dy)
#pragma unroll
            for (int dx = 0; dx < 4; ++dx)
                sum += xb[((size_t)dy * 304 + dx) * TSTEPS + t];
        u = u * I_DECAY + POOL_WT * sum;
        v = v * V_DECAY + u;
        float s = (v >= THETA) ? 1.f : 0.f;
        v *= (1.f - s);
        if (t < TSTEPS - 1) sb[(size_t)(t + 1) * CHW] = (_Float16)s;
    }
}

// ---------------------------------------------------------------------------
// 2x2 sum-pool on spike maps [N,T,C,H,W] fused with CUBA scan + delay.
// ---------------------------------------------------------------------------
__global__ void pool2_loihi_kernel(const _Float16* __restrict__ X,
                                   _Float16* __restrict__ S,
                                   int C, int H, int W, int total) {
    int nid = blockIdx.x * blockDim.x + threadIdx.x;
    if (nid >= total) return;
    int Ho = H >> 1, Wo = W >> 1;
    int w = nid % Wo, h = (nid / Wo) % Ho, c = (nid / (Wo * Ho)) % C, n = nid / (Wo * Ho * C);
    const int CHWi = C * H * W, CHWo = C * Ho * Wo;
    const _Float16* xb = X + (size_t)n * TSTEPS * CHWi + (size_t)c * H * W + (2 * h) * W + 2 * w;
    _Float16* sb = S + (size_t)n * TSTEPS * CHWo + (size_t)c * Ho * Wo + h * Wo + w;
    sb[0] = (_Float16)0.f;
    float u = 0.f, v = 0.f;
    for (int t = 0; t < TSTEPS; ++t) {
        const _Float16* xt = xb + (size_t)t * CHWi;
        float sum = (float)xt[0] + (float)xt[1] + (float)xt[W] + (float)xt[W + 1];
        u = u * I_DECAY + POOL_WT * sum;
        v = v * V_DECAY + u;
        float s = (v >= THETA) ? 1.f : 0.f;
        v *= (1.f - s);
        if (t < TSTEPS - 1) sb[(size_t)(t + 1) * CHWo] = (_Float16)s;
    }
}

// ---------------------------------------------------------------------------
// CUBA neuron scan + delay on f32 pre-activations, layout [N,T,F] -> f16 spikes
// ---------------------------------------------------------------------------
__global__ void loihi_kernel(const float* __restrict__ X,
                             _Float16* __restrict__ S,
                             int F, int total) {
    int nid = blockIdx.x * blockDim.x + threadIdx.x;
    if (nid >= total) return;
    int f = nid % F, n = nid / F;
    const float* xb = X + (size_t)n * TSTEPS * F + f;
    _Float16* sb = S + (size_t)n * TSTEPS * F + f;
    sb[0] = (_Float16)0.f;
    float u = 0.f, v = 0.f;
    for (int t = 0; t < TSTEPS; ++t) {
        u = u * I_DECAY + xb[(size_t)t * F];
        v = v * V_DECAY + u;
        float s = (v >= THETA) ? 1.f : 0.f;
        v *= (1.f - s);
        if (t < TSTEPS - 1) sb[(size_t)(t + 1) * F] = (_Float16)s;
    }
}

// ---------------------------------------------------------------------------
// f32 -> f16 weight conversion (grid-stride).
// ---------------------------------------------------------------------------
__global__ void f32_to_f16_kernel(const float* __restrict__ X,
                                  _Float16* __restrict__ Y, int n) {
    int stride = gridDim.x * blockDim.x;
    for (int i = blockIdx.x * blockDim.x + threadIdx.x; i < n; i += stride)
        Y[i] = (_Float16)X[i];
}

// ---------------------------------------------------------------------------
// Implicit-GEMM conv per (n,t) image via v_wmma_f32_16x16x32_f16.
// Compile-time dims -> div/mod fold to mul-shift, K-loop fully unrolls.
// One wave32 per 16(outch) x 16(positions) tile. Out-of-range position lanes
// are clamped (their D columns are never stored), so EXEC stays all-1s.
// ---------------------------------------------------------------------------
template <int CIN, int O, int H, int W, int KSZ, int KEFF>
__global__ void conv_wmma_kernel(const _Float16* __restrict__ Sin, // [N,T,CIN,H,W]
                                 const float* __restrict__ Wt,     // [O, CIN*KSZ*KSZ]
                                 float* __restrict__ Out) {        // [N,T,O,H,W]
    constexpr int HW  = H * W;
    constexpr int KK2 = KSZ * KSZ;
    constexpr int PAD = KSZ / 2;
    const int lane = threadIdx.x;
    const int img  = blockIdx.z;                      // (n,t) flat image index
    int pos = blockIdx.x * 16 + (lane & 15);          // N (column) role of lane
    if (pos >= HW) pos = HW - 1;                      // clamp; col never stored
    const int h = pos / W, w = pos % W;
    const int m = blockIdx.y * 16 + (lane & 15);      // M (row) role of lane
    const _Float16* simg = Sin + (size_t)img * CIN * HW;

    v8f acc = {};
#pragma unroll
    for (int k0 = 0; k0 < KEFF; k0 += 32) {
        v16h a, b;
#pragma unroll
        for (int i = 0; i < 16; ++i) {
            int k = k0 + kmap(lane, i);
            a[i] = (_Float16)((k < KEFF) ? Wt[m * KEFF + k] : 0.f);
            _Float16 bv = (_Float16)0.f;
            if (k < KEFF) {
                int c  = k / KK2, r = k % KK2;
                int hh = h + r / KSZ - PAD;
                int ww = w + r % KSZ - PAD;
                if (hh >= 0 && hh < H && ww >= 0 && ww < W)
                    bv = simg[c * HW + hh * W + ww];
            }
            b[i] = bv;
        }
        acc = __builtin_amdgcn_wmma_f32_16x16x32_f16(false, a, false, b,
                                                     (short)0, acc, false, false);
    }
    // C/D layout: VGPR v, lanes0-15 -> M=v, lanes16-31 -> M=v+8; N = lane&15
    const int half = lane >> 4;
    const int col  = blockIdx.x * 16 + (lane & 15);
    if (col < HW) {
        float* oimg = Out + ((size_t)img * O + blockIdx.y * 16) * HW;
#pragma unroll
        for (int vg = 0; vg < 8; ++vg)
            oimg[(size_t)(vg + half * 8) * HW + col] = acc[vg];
    }
}

// ---------------------------------------------------------------------------
// fc1 GEMM: [M=512 x K=15200](f16) x [K x NT=100](f16 spikes) via WMMA.
// Fragment K-map is contiguous in runs of 8 per lane, so each A/B fragment is
// exactly two b128 loads. Padding columns are clamped, not masked.
// ---------------------------------------------------------------------------
__global__ void fc_wmma_kernel(const _Float16* __restrict__ Sin, // [NT, K]
                               const _Float16* __restrict__ Wh,  // [M, K] f16
                               float* __restrict__ Out,          // [NT, M]
                               int K, int M, int NT) {
    const int lane = threadIdx.x;
    const int klo  = (lane & 16) >> 1;                 // 0 or 8
    const int m    = blockIdx.y * 16 + (lane & 15);
    int col = blockIdx.x * 16 + (lane & 15);
    if (col >= NT) col = NT - 1;                       // clamp; never stored
    const _Float16* ap = Wh  + (size_t)m   * K + klo;
    const _Float16* bp = Sin + (size_t)col * K + klo;

    v8f acc = {};
    for (int k0 = 0; k0 < K; k0 += 32) {
        __builtin_prefetch(bp + k0 + 256, 0, 1);       // global_prefetch_b8
        v8h a0 = *(const v8h*)(ap + k0);
        v8h a1 = *(const v8h*)(ap + k0 + 16);
        v8h b0 = *(const v8h*)(bp + k0);
        v8h b1 = *(const v8h*)(bp + k0 + 16);
        v16h a, b;
#pragma unroll
        for (int i = 0; i < 8; ++i) {
            a[i] = a0[i]; a[8 + i] = a1[i];
            b[i] = b0[i]; b[8 + i] = b1[i];
        }
        acc = __builtin_amdgcn_wmma_f32_16x16x32_f16(false, a, false, b,
                                                     (short)0, acc, false, false);
    }
    const int half = lane >> 4;
    const int c2   = blockIdx.x * 16 + (lane & 15);
    if (c2 < NT) {
        float* ob = Out + (size_t)c2 * M + blockIdx.y * 16 + half * 8;
        v4f lo = {acc[0], acc[1], acc[2], acc[3]};
        v4f hi = {acc[4], acc[5], acc[6], acc[7]};
        *(v4f*)ob       = lo;
        *(v4f*)(ob + 4) = hi;
    }
}

// ---------------------------------------------------------------------------
// fc2 (512 -> 2) fused with the final CUBA scan + delay; writes d_out
// [N,2,1,1,T] float. Only 4 neurons total.
// ---------------------------------------------------------------------------
__global__ void fc2_loihi_kernel(const _Float16* __restrict__ S6, // [N,T,512]
                                 const float* __restrict__ W2,    // [2,512]
                                 float* __restrict__ Out) {
    int tid = threadIdx.x;
    if (tid >= 4) return;
    int n = tid >> 1, c = tid & 1;
    float u = 0.f, v = 0.f;
    Out[(n * 2 + c) * TSTEPS + 0] = 0.f;
    for (int t = 0; t < TSTEPS; ++t) {
        const _Float16* sb = S6 + ((size_t)n * TSTEPS + t) * 512;
        const float* wc = W2 + c * 512;
        float x = 0.f;
        for (int j = 0; j < 512; j += 8) {
            v8h sv = *(const v8h*)(sb + j);
#pragma unroll
            for (int q = 0; q < 8; ++q) x += wc[j + q] * (float)sv[q];
        }
        u = u * I_DECAY + x;
        v = v * V_DECAY + u;
        float s = (v >= THETA) ? 1.f : 0.f;
        v *= (1.f - s);
        if (t < TSTEPS - 1) Out[(n * 2 + c) * TSTEPS + t + 1] = s;
    }
}

// ---------------------------------------------------------------------------
// Workspace layout (bytes). BIG is aliased for conv1/conv2 pre-activations;
// S2 region is aliased for s4. Peak ~101 MB.
// ---------------------------------------------------------------------------
static const size_t OFF_S1     = 0;            //  1,520,000 f16  [2,50, 2,100,76]
static const size_t OFF_BIG    = 3040000;      // 12,160,000 f32  conv pre-acts
static const size_t OFF_S2     = 51680000;     // 12,160,000 f16  [2,50,16,100,76]
static const size_t OFF_S3     = 76000000;     //  3,040,000 f16  [2,50,16, 50,38]
static const size_t OFF_S5     = 82080000;     //  1,520,000 f16  [2,50,32, 25,19]
static const size_t OFF_FC1PRE = 85120000;     //    204,800 B f32 [2,50,512]
static const size_t OFF_S6     = 85324800;     //    102,400 B f16 [2,50,512]
static const size_t OFF_FC1WH  = 85427200;     // 15,564,800 B f16 [512,15200]

extern "C" void kernel_launch(void* const* d_in, const int* in_sizes, int n_in,
                              void* d_out, int out_size, void* d_ws, size_t ws_size,
                              hipStream_t stream) {
    const float* spike_in = (const float*)d_in[0];   // [2,2,400,304,50]
    const float* conv1_w  = (const float*)d_in[1];   // [16,2,5,5]
    const float* conv2_w  = (const float*)d_in[2];   // [32,16,3,3]
    const float* fc1_w    = (const float*)d_in[3];   // [512,15200]
    const float* fc2_w    = (const float*)d_in[4];   // [2,512]
    float* out = (float*)d_out;
    char*  ws  = (char*)d_ws;

    _Float16* s1     = (_Float16*)(ws + OFF_S1);
    float*    c1pre  = (float*)   (ws + OFF_BIG);
    _Float16* s2     = (_Float16*)(ws + OFF_S2);
    _Float16* s3     = (_Float16*)(ws + OFF_S3);
    float*    c2pre  = c1pre;          // alias: c1pre dead before conv2
    _Float16* s4     = s2;             // alias: s2 dead before its CUBA scan
    _Float16* s5     = (_Float16*)(ws + OFF_S5);
    float*    fc1pre = (float*)   (ws + OFF_FC1PRE);
    _Float16* s6     = (_Float16*)(ws + OFF_S6);
    _Float16* fc1wh  = (_Float16*)(ws + OFF_FC1WH);

    // Pre-convert fc1 weights to f16 (done early; independent of pipeline).
    f32_to_f16_kernel<<<2048, 256, 0, stream>>>(fc1_w, fc1wh, 512 * 15200);

    // A: pool4 + spike + delay      [2,2,400,304,50] -> s1 [2,50,2,100,76]
    pool4_loihi_kernel<<<(30400 + 255) / 256, 256, 0, stream>>>(spike_in, s1);

    // B: conv1 5x5, 2->16 (WMMA implicit GEMM, K=50 pad 64)
    conv_wmma_kernel<2, 16, 100, 76, 5, 50><<<dim3(475, 1, 100), 32, 0, stream>>>(
        s1, conv1_w, c1pre);

    // C: spike + delay on conv1 out -> s2
    loihi_kernel<<<(243200 + 255) / 256, 256, 0, stream>>>(c1pre, s2, 16 * 7600, 243200);

    // D: pool2 + spike + delay -> s3 [2,50,16,50,38]
    pool2_loihi_kernel<<<(60800 + 255) / 256, 256, 0, stream>>>(s2, s3, 16, 100, 76, 60800);

    // E: conv2 3x3, 16->32 (WMMA, K=144 pad 160)
    conv_wmma_kernel<16, 32, 50, 38, 3, 144><<<dim3(119, 2, 100), 32, 0, stream>>>(
        s3, conv2_w, c2pre);

    // F: spike + delay -> s4
    loihi_kernel<<<(121600 + 255) / 256, 256, 0, stream>>>(c2pre, s4, 32 * 1900, 121600);

    // G: pool2 + spike + delay -> s5 [2,50,32,25,19] == fc1 columns [100,15200]
    pool2_loihi_kernel<<<(30400 + 255) / 256, 256, 0, stream>>>(s4, s5, 32, 50, 38, 30400);

    // H: fc1 GEMM 512x15200 @ 15200x100 (WMMA), then spike + delay -> s6
    fc_wmma_kernel<<<dim3(7, 32, 1), 32, 0, stream>>>(s5, fc1wh, fc1pre, 15200, 512, 100);
    loihi_kernel<<<(1024 + 255) / 256, 256, 0, stream>>>(fc1pre, s6, 512, 1024);

    // I: fc2 + spike + delay -> d_out [2,2,1,1,50]
    fc2_loihi_kernel<<<1, 32, 0, stream>>>(s6, fc2_w, out);
}
